// ContextualLoss_72670846648945
// MI455X (gfx1250) — matrix operand order repaired and checked by reading, not compiled, not run on hardware
//
#include <hip/hip_runtime.h>
#include <hip/hip_bf16.h>

typedef _Float16 v8h  __attribute__((ext_vector_type(8)));
typedef _Float16 v16h __attribute__((ext_vector_type(16)));
typedef float    v8f  __attribute__((ext_vector_type(8)));

#define N_PIX   4096
#define CCH     64
#define NBATCH  4
#define EPS_MIN 1e-5f
#define ROWT    2                       // 16-row tiles per block (32 rows/block)
#define C_LOG2E 14.4269504088896341f    // (1/SIGMA) * log2(e)

// ---------------- Stage 1: channel-wise L2 normalize, transpose to [N, C] f16 ----------------
__global__ void cx_normalize_kernel(const float* __restrict__ src,
                                    _Float16* __restrict__ dst,
                                    float* __restrict__ sq) {
    int idx = blockIdx.x * blockDim.x + threadIdx.x;       // 0 .. B*N-1
    int b = idx >> 12;
    int n = idx & (N_PIX - 1);
    const float* p = src + (size_t)b * CCH * N_PIX + n;
    float vals[CCH];
    float ss = 0.0f;
#pragma unroll
    for (int c = 0; c < CCH; ++c) {
        float v = p[(size_t)c * N_PIX];                    // coalesced across lanes
        vals[c] = v;
        ss += v * v;
    }
    float inv = 1.0f / fmaxf(sqrtf(ss), 1e-12f);
    _Float16* q = dst + (size_t)idx * CCH;
#pragma unroll
    for (int c = 0; c < CCH; ++c) q[c] = (_Float16)(vals[c] * inv);
    sq[idx] = ss * inv * inv;                              // ||xn||^2
}

// ---------------- WMMA fragment loaders (CDNA5 wave32 layouts, 16-bit) ----------------
__device__ inline v16h load_a_frag(const _Float16* __restrict__ base, int n0, int kbase, int lane) {
    int row  = n0 + (lane & 15);
    int ksub = (lane >> 4) * 8;
    const _Float16* p = base + (size_t)row * CCH + kbase + ksub;
    v8h lo = *(const v8h*)(p);
    v8h hi = *(const v8h*)(p + 16);
    v16h r;
#pragma unroll
    for (int i = 0; i < 8; ++i) { r[i] = lo[i]; r[i + 8] = hi[i]; }
    return r;
}

__device__ inline v16h load_b_frag(const _Float16* __restrict__ base, int m0, int kbase, int lane) {
    int col  = m0 + (lane & 15);
    int ksub = (lane >> 4) * 8;
    const _Float16* p = base + (size_t)col * CCH + kbase + ksub;
    v8h lo = *(const v8h*)(p);
    v8h hi = *(const v8h*)(p + 16);
    v16h r;
#pragma unroll
    for (int i = 0; i < 8; ++i) { r[i] = lo[i]; r[i + 8] = hi[i]; }
    return r;
}

__device__ inline v8f wmma_k64(v16h a0, v16h a1, v16h b0, v16h b1) {
    v8f c = {};
    c = __builtin_amdgcn_wmma_f32_16x16x32_f16(false, a0, false, b0, (short)0, c, false, false);
    c = __builtin_amdgcn_wmma_f32_16x16x32_f16(false, a1, false, b1, (short)0, c, false, false);
    return c;
}

// ---------------- Pass 1: per-row min of (y2 - 2*dot); x2 and sqrt folded into finalize --------
// Block = 256 thr = 8 waves; block owns 32 rows (2 tiles); each wave handles 2 column tiles/iter.
__global__ void cx_min_kernel(const _Float16* __restrict__ xn, const _Float16* __restrict__ yn,
                              const float* __restrict__ x2, const float* __restrict__ y2,
                              float* __restrict__ dmin) {
    __shared__ float part[8][32];
    int b  = blockIdx.x >> 7;
    int n0 = (blockIdx.x & 127) * 32;
    int lane = threadIdx.x & 31, w = threadIdx.x >> 5;
    const _Float16* xb = xn + (size_t)b * N_PIX * CCH;
    const _Float16* yb = yn + (size_t)b * N_PIX * CCH;
    const float* y2b = y2 + b * N_PIX;

    v16h a0[ROWT], a1[ROWT];
    float dl[ROWT][8];
#pragma unroll
    for (int rt = 0; rt < ROWT; ++rt) {
        a0[rt] = load_a_frag(xb, n0 + rt * 16, 0, lane);
        a1[rt] = load_a_frag(xb, n0 + rt * 16, 32, lane);
#pragma unroll
        for (int j = 0; j < 8; ++j) dl[rt][j] = 3.4e38f;
    }

    for (int m0 = w * 32; m0 < N_PIX; m0 += 256) {
        v16h b0a = load_b_frag(yb, m0, 0, lane);
        v16h b1a = load_b_frag(yb, m0, 32, lane);
        v16h b0b = load_b_frag(yb, m0 + 16, 0, lane);
        v16h b1b = load_b_frag(yb, m0 + 16, 32, lane);
        float cy2a = y2b[m0 + (lane & 15)];
        float cy2b = y2b[m0 + 16 + (lane & 15)];
#pragma unroll
        for (int rt = 0; rt < ROWT; ++rt) {
            v8f ca = wmma_k64(a0[rt], a1[rt], b0a, b1a);
            v8f cb = wmma_k64(a0[rt], a1[rt], b0b, b1b);
#pragma unroll
            for (int j = 0; j < 8; ++j) {
                float ea = fmaf(-2.0f, ca[j], cy2a);       // y2 - 2*dot (x2 added at finalize)
                float eb = fmaf(-2.0f, cb[j], cy2b);
                dl[rt][j] = fminf(dl[rt][j], fminf(ea, eb));   // folds to v_min3_num_f32
            }
        }
    }
#pragma unroll
    for (int m = 1; m <= 8; m <<= 1)
#pragma unroll
        for (int rt = 0; rt < ROWT; ++rt)
#pragma unroll
            for (int j = 0; j < 8; ++j)
                dl[rt][j] = fminf(dl[rt][j], __shfl_xor(dl[rt][j], m, 32));
    if ((lane & 15) == 0) {
        int h = lane >> 4;
#pragma unroll
        for (int rt = 0; rt < ROWT; ++rt)
#pragma unroll
            for (int j = 0; j < 8; ++j) part[w][rt * 16 + j + 8 * h] = dl[rt][j];
    }
    __syncthreads();
    if (threadIdx.x < 32) {
        float v = part[0][threadIdx.x];
#pragma unroll
        for (int ww = 1; ww < 8; ++ww) v = fminf(v, part[ww][threadIdx.x]);
        int row = b * N_PIX + n0 + threadIdx.x;
        float d2 = x2[row] + v;                            // add back ||xn||^2
        dmin[row] = __builtin_amdgcn_sqrtf(fmaxf(d2, 0.0f));
    }
}

// ---------------- Pass 2: per-row sum of w = exp2(C - d*rs); emit cx_max ----------------
__global__ void cx_sum_kernel(const _Float16* __restrict__ xn, const _Float16* __restrict__ yn,
                              const float* __restrict__ x2, const float* __restrict__ y2,
                              const float* __restrict__ dmin, float* __restrict__ cxmax) {
    __shared__ float part[8][32];
    int b  = blockIdx.x >> 7;
    int n0 = (blockIdx.x & 127) * 32;
    int lane = threadIdx.x & 31, w = threadIdx.x >> 5;
    const _Float16* xb = xn + (size_t)b * N_PIX * CCH;
    const _Float16* yb = yn + (size_t)b * N_PIX * CCH;
    const float* y2b = y2 + b * N_PIX;

    v16h a0[ROWT], a1[ROWT];
    float rx2[ROWT][8], rs[ROWT][8], ssum[ROWT][8];
#pragma unroll
    for (int rt = 0; rt < ROWT; ++rt) {
        a0[rt] = load_a_frag(xb, n0 + rt * 16, 0, lane);
        a1[rt] = load_a_frag(xb, n0 + rt * 16, 32, lane);
        int rbase = b * N_PIX + n0 + rt * 16 + 8 * (lane >> 4);
#pragma unroll
        for (int j = 0; j < 8; ++j) {
            rx2[rt][j]  = x2[rbase + j];
            rs[rt][j]   = C_LOG2E / (dmin[rbase + j] + EPS_MIN);
            ssum[rt][j] = 0.0f;
        }
    }

    for (int m0 = w * 32; m0 < N_PIX; m0 += 256) {
        v16h b0a = load_b_frag(yb, m0, 0, lane);
        v16h b1a = load_b_frag(yb, m0, 32, lane);
        v16h b0b = load_b_frag(yb, m0 + 16, 0, lane);
        v16h b1b = load_b_frag(yb, m0 + 16, 32, lane);
        float cy2a = y2b[m0 + (lane & 15)];
        float cy2b = y2b[m0 + 16 + (lane & 15)];
#pragma unroll
        for (int rt = 0; rt < ROWT; ++rt) {
            v8f ca = wmma_k64(a0[rt], a1[rt], b0a, b1a);
            v8f cb = wmma_k64(a0[rt], a1[rt], b0b, b1b);
#pragma unroll
            for (int j = 0; j < 8; ++j) {
                float d2a = rx2[rt][j] + fmaf(-2.0f, ca[j], cy2a);
                float d2b = rx2[rt][j] + fmaf(-2.0f, cb[j], cy2b);
                float da  = __builtin_amdgcn_sqrtf(fmaxf(d2a, 0.0f));   // single v_sqrt_f32
                float db  = __builtin_amdgcn_sqrtf(fmaxf(d2b, 0.0f));
                float wa  = __builtin_amdgcn_exp2f(fmaf(-da, rs[rt][j], C_LOG2E));
                float wb  = __builtin_amdgcn_exp2f(fmaf(-db, rs[rt][j], C_LOG2E));
                ssum[rt][j] += (wa + wb);
            }
        }
    }
#pragma unroll
    for (int m = 1; m <= 8; m <<= 1)
#pragma unroll
        for (int rt = 0; rt < ROWT; ++rt)
#pragma unroll
            for (int j = 0; j < 8; ++j)
                ssum[rt][j] += __shfl_xor(ssum[rt][j], m, 32);
    if ((lane & 15) == 0) {
        int h = lane >> 4;
#pragma unroll
        for (int rt = 0; rt < ROWT; ++rt)
#pragma unroll
            for (int j = 0; j < 8; ++j) part[w][rt * 16 + j + 8 * h] = ssum[rt][j];
    }
    __syncthreads();
    if (threadIdx.x < 32) {
        float s = 0.0f;
#pragma unroll
        for (int ww = 0; ww < 8; ++ww) s += part[ww][threadIdx.x];
        int row = b * N_PIX + n0 + threadIdx.x;
        float dm = dmin[row];
        float wmax = __builtin_amdgcn_exp2f(C_LOG2E * (1.0f - dm / (dm + EPS_MIN)));
        cxmax[row] = wmax / s;
    }
}

// ---------------- Pass 3: -log(mean(cx_max + eps)) ----------------
__global__ void cx_final_kernel(const float* __restrict__ cxmax, float* __restrict__ out, int total) {
    __shared__ float sm[256];
    float s = 0.0f;
    for (int i = threadIdx.x; i < total; i += 256) s += cxmax[i] + EPS_MIN;
    sm[threadIdx.x] = s;
    __syncthreads();
    for (int off = 128; off > 0; off >>= 1) {
        if (threadIdx.x < off) sm[threadIdx.x] += sm[threadIdx.x + off];
        __syncthreads();
    }
    if (threadIdx.x == 0) out[0] = -logf(sm[0] / (float)total);
}

extern "C" void kernel_launch(void* const* d_in, const int* in_sizes, int n_in,
                              void* d_out, int out_size, void* d_ws, size_t ws_size,
                              hipStream_t stream) {
    const float* x = (const float*)d_in[0];
    const float* y = (const float*)d_in[1];
    float* out = (float*)d_out;

    char* ws = (char*)d_ws;
    const size_t nvec = (size_t)NBATCH * N_PIX;            // 16384 rows
    _Float16* xn = (_Float16*)ws;                          // 2 MB
    _Float16* yn = xn + nvec * CCH;                        // 2 MB
    float* x2    = (float*)(ws + 2 * nvec * CCH * sizeof(_Float16));
    float* y2    = x2 + nvec;
    float* dmin  = y2 + nvec;
    float* cxmax = dmin + nvec;

    cx_normalize_kernel<<<nvec / 256, 256, 0, stream>>>(x, xn, x2);
    cx_normalize_kernel<<<nvec / 256, 256, 0, stream>>>(y, yn, y2);
    cx_min_kernel<<<NBATCH * (N_PIX / 32), 256, 0, stream>>>(xn, yn, x2, y2, dmin);
    cx_sum_kernel<<<NBATCH * (N_PIX / 32), 256, 0, stream>>>(xn, yn, x2, y2, dmin, cxmax);
    cx_final_kernel<<<1, 256, 0, stream>>>(cxmax, out, (int)nvec);
}
